// TransformerEncoderLayer_90125593739687
// MI455X (gfx1250) — compile-verified, matrix operand order checked by reading, not run on hardware
//
// MI455X (gfx1250, CDNA5) transformer encoder layer: MHA + top-2 MoE FFN.
// Wave32, WMMA bf16 16x16x32, TDM async tensor loads, block-skip MoE sparsity.
#include <hip/hip_runtime.h>
#include <hip/hip_bf16.h>
#include <cstdint>
#include <cstddef>

// ---------------- problem constants ----------------
constexpr int B_    = 4;
constexpr int S_    = 2048;
constexpr int D_    = 1024;
constexpr int H_    = 16;
constexpr int F_    = 4096;
constexpr int E_    = 8;
constexpr int KTOP_ = 2;
constexpr int HD_   = D_ / H_;   // 64
constexpr int T_    = B_ * S_;   // 8192
constexpr float EPS_ = 1e-5f;

typedef __bf16 bf16;
typedef __attribute__((ext_vector_type(16))) __bf16 v16bf;
typedef __attribute__((ext_vector_type(8)))  __bf16 v8bf;
typedef __attribute__((ext_vector_type(4)))  __bf16 v4bf;
typedef __attribute__((ext_vector_type(8)))  float  v8f;
typedef __attribute__((ext_vector_type(4)))  unsigned int u32x4;
typedef __attribute__((ext_vector_type(8)))  int i32x8;
typedef __attribute__((ext_vector_type(4)))  int i32x4;

// ---------------- CDNA5 feature detection ----------------
#if __has_builtin(__builtin_amdgcn_tensor_load_to_lds)
#define USE_TDM 1
#else
#define USE_TDM 0
#endif

#if __has_builtin(__builtin_amdgcn_s_wait_tensorcnt)
#define WAIT_TENSORCNT0() __builtin_amdgcn_s_wait_tensorcnt(0)
#else
#define WAIT_TENSORCNT0() asm volatile("s_wait_tensorcnt 0x0" ::: "memory")
#endif

// ---------------- small helpers ----------------
__device__ __forceinline__ v8f zero_v8f() {
  v8f z;
#pragma unroll
  for (int i = 0; i < 8; ++i) z[i] = 0.0f;
  return z;
}

__device__ __forceinline__ v8f wmma_bf16(v16bf a, v16bf b, v8f c) {
  // D = A(16x32) * B(32x16) + C(16x16 f32)
  return __builtin_amdgcn_wmma_f32_16x16x32_bf16(false, a, false, b, (short)0, c,
                                                 false, false);
}

// CDNA5 16-bit A/B fragment VGPR layout (ISA 7.12.2): lane l -> outer = l%16,
// halves 0..7 hold K = (l/16)*8 + 0..7, halves 8..15 hold K = 16 + (l/16)*8 + 0..7.
// LDS tile is [outer][K] row-major with `stride` bf16 elements per row; both
// 8-element runs are 16B aligned so this lowers to two ds_load_b128.
__device__ __forceinline__ v16bf load_frag(const bf16* base, int lo, int ksel,
                                           int stride) {
  const bf16* p = base + lo * stride + ksel * 8;
  v8bf l = *(const v8bf*)p;
  v8bf h = *(const v8bf*)(p + 16);
  v16bf r;
#pragma unroll
  for (int i = 0; i < 8; ++i) { r[i] = l[i]; r[i + 8] = h[i]; }
  return r;
}

__device__ __forceinline__ float gelu_exact(float v) {
  return 0.5f * v * (1.0f + erff(v * 0.70710678118654752f));
}

// ---------------- TDM: 2-D f32 tile -> LDS ----------------
#if USE_TDM
typedef char __attribute__((address_space(3))) lds_char_t;
__device__ __forceinline__ unsigned lds_offset_of(const void* p) {
  return (unsigned)(unsigned long long)(lds_char_t*)p;  // ptrtoint of as(3)
}

#define RFL(x) __builtin_amdgcn_readfirstlane((int)(x))

__device__ __forceinline__ void tdm_load_f32_tile(unsigned lds_off, const float* g,
                                                  int tile_x, int tile_y,
                                                  long long row_stride,
                                                  long long tensor_x,
                                                  long long tensor_y) {
  unsigned long long ga = (unsigned long long)g;
  u32x4 g0;
  g0[0] = (unsigned)RFL(1u);                                     // count=1, user D#
  g0[1] = (unsigned)RFL(lds_off);                                // lds_addr
  g0[2] = (unsigned)RFL(ga & 0xffffffffull);                     // global_addr lo
  g0[3] = (unsigned)RFL(((ga >> 32) & 0x1ffffffull) | (2u << 30)); // hi | type=2
  i32x8 g1;
  g1[0] = RFL(2u << 16);                                         // data_size=4B
  g1[1] = RFL(((unsigned)tensor_x & 0xffffu) << 16);             // tdim0[15:0]
  g1[2] = RFL((((unsigned)tensor_x >> 16) & 0xffffu) |
              (((unsigned)tensor_y & 0xffffu) << 16));           // tdim0 hi | tdim1 lo
  g1[3] = RFL((((unsigned)tensor_y >> 16) & 0xffffu) |
              (((unsigned)tile_x & 0xffffu) << 16));             // tdim1 hi | tile0
  g1[4] = RFL((unsigned)tile_y & 0xffffu);                       // tile1 (tile2=0)
  g1[5] = RFL((unsigned)(row_stride & 0xffffffffll));            // dim0_stride lo
  g1[6] = RFL((unsigned)((row_stride >> 32) & 0xffffll));        // dim0_stride hi
  g1[7] = 0;
  i32x4 z4; z4[0] = z4[1] = z4[2] = z4[3] = 0;
#if defined(__clang_major__) && (__clang_major__ >= 23)
  i32x8 z8;
#pragma unroll
  for (int i = 0; i < 8; ++i) z8[i] = 0;
  __builtin_amdgcn_tensor_load_to_lds(g0, g1, z4, z4, z8, 0);
#else
  __builtin_amdgcn_tensor_load_to_lds(g0, g1, z4, z4, 0);
#endif
}
#endif  // USE_TDM

// ================= generic WMMA GEMM: C = op(A[MxK] @ W[KxN] + bias) =========
// MODE 0: C = A@W + b           MODE 1: C = gelu(A@W + b)  (+ block skip)
// MODE 2: C += rowscale * (A@W + b)  (block skip on all-zero rowscale tile)
#define BM 128
#define BN 128
#define BK 32
#define ASTR 40  // padded bf16 stride for LDS [outer][K] tiles (bank spread)

template <int MODE>
__global__ void __launch_bounds__(256)
gemm_kernel(const float* __restrict__ A, const float* __restrict__ W,
            const float* __restrict__ bias, float* __restrict__ C,
            int M, int N, int Kd,
            const float* __restrict__ rowscale, int rsStride) {
  __shared__ __attribute__((aligned(16))) bf16 As[BM * ASTR];
  __shared__ __attribute__((aligned(16))) bf16 Bs[BN * ASTR];
#if USE_TDM
  __shared__ __attribute__((aligned(16))) float Astage[BM * BK];
#endif
  __shared__ float s_rs[BM];
  __shared__ int s_active;

  const int tid  = threadIdx.x;
  const int lane = tid & 31, wave = tid >> 5;
  const int wm = wave & 3, wn = wave >> 2;          // 4 x 2 wave grid
  const int lo = lane & 15, ksel = lane >> 4;
  const int row0 = blockIdx.y * BM, col0 = blockIdx.x * BN;

  // ---- top-k sparsity: skip tile if all 128 routing weights are zero ----
  if (rowscale) {
    if (tid == 0) s_active = 0;
    __syncthreads();
    if (tid < BM) {
      float wv = rowscale[(size_t)(row0 + tid) * rsStride];
      s_rs[tid] = wv;
      if (wv != 0.0f) atomicOr(&s_active, 1);
    }
    __syncthreads();
    if (!s_active) return;
  }

  v8f acc[2][4];
#pragma unroll
  for (int i = 0; i < 2; ++i)
#pragma unroll
    for (int j = 0; j < 4; ++j) acc[i][j] = zero_v8f();

  for (int k0 = 0; k0 < Kd; k0 += BK) {
    __syncthreads();  // previous iteration's fragment reads done

#if USE_TDM
    if (wave == 0) {  // one async tensor DMA per block: A tile -> LDS staging
      tdm_load_f32_tile(lds_offset_of(&Astage[0]),
                        A + (size_t)row0 * Kd + k0, BK, BM, (long long)Kd,
                        (long long)(Kd - k0), (long long)(M - row0));
      WAIT_TENSORCNT0();
    }
#endif
    // W tile (BK x BN) -> Bs[n][k] (transposed bf16), overlaps the TDM
    for (int idx = tid; idx < BK * BN / 4; idx += 256) {
      int kk = idx >> 5, n4 = idx & 31;
      const float4 v = *(const float4*)(W + (size_t)(k0 + kk) * N + col0 + n4 * 4);
      Bs[(n4 * 4 + 0) * ASTR + kk] = (bf16)v.x;
      Bs[(n4 * 4 + 1) * ASTR + kk] = (bf16)v.y;
      Bs[(n4 * 4 + 2) * ASTR + kk] = (bf16)v.z;
      Bs[(n4 * 4 + 3) * ASTR + kk] = (bf16)v.w;
    }
#if USE_TDM
    __syncthreads();  // Astage ready
    for (int idx = tid; idx < BM * BK / 4; idx += 256) {
      int r = idx >> 3, c4 = idx & 7;
      const float4 v = *(const float4*)(Astage + r * BK + c4 * 4);
      v4bf o = {(bf16)v.x, (bf16)v.y, (bf16)v.z, (bf16)v.w};
      *(v4bf*)&As[r * ASTR + c4 * 4] = o;
    }
#else
    for (int idx = tid; idx < BM * BK / 4; idx += 256) {
      int r = idx >> 3, c4 = idx & 7;
      const float4 v = *(const float4*)(A + (size_t)(row0 + r) * Kd + k0 + c4 * 4);
      v4bf o = {(bf16)v.x, (bf16)v.y, (bf16)v.z, (bf16)v.w};
      *(v4bf*)&As[r * ASTR + c4 * 4] = o;
    }
#endif
    if (k0 + BK < Kd) {  // stream-ahead hint (global_prefetch_b8)
      __builtin_prefetch(W + (size_t)(k0 + BK + (tid >> 5)) * N + col0 + (tid & 31) * 4, 0, 1);
    }
    __syncthreads();  // As & Bs ready

    v16bf afrag[2], bfrag[4];
#pragma unroll
    for (int i = 0; i < 2; ++i)
      afrag[i] = load_frag(&As[(wm * 32 + i * 16) * ASTR], lo, ksel, ASTR);
#pragma unroll
    for (int j = 0; j < 4; ++j)
      bfrag[j] = load_frag(&Bs[(wn * 64 + j * 16) * ASTR], lo, ksel, ASTR);
#pragma unroll
    for (int i = 0; i < 2; ++i)
#pragma unroll
      for (int j = 0; j < 4; ++j)
        acc[i][j] = wmma_bf16(afrag[i], bfrag[j], acc[i][j]);
  }

  // ---- epilogue (C-matrix layout: row = i*16 + ii + 8*(lane/16), col = lo) ----
#pragma unroll
  for (int i = 0; i < 2; ++i) {
    const int rbase = row0 + wm * 32 + i * 16 + 8 * ksel;
#pragma unroll
    for (int j = 0; j < 4; ++j) {
      const int col = col0 + wn * 64 + j * 16 + lo;
      const float bv = bias ? bias[col] : 0.0f;
#pragma unroll
      for (int ii = 0; ii < 8; ++ii) {
        const int row = rbase + ii;
        float v = acc[i][j][ii] + bv;
        if constexpr (MODE == 1) v = gelu_exact(v);
        const size_t o = (size_t)row * N + col;
        if constexpr (MODE == 2)
          C[o] += s_rs[row - row0] * v;
        else
          C[o] = v;
      }
    }
  }
}

// ================= flash attention: per (b, h, 128-query block) ==============
#define QB 128
#define KB 128
#define QKSTR 72   // bf16 [row][64]
#define VSTR  136  // bf16 Vs[d][key]
#define SSTR  132  // f32 scores
#define PSTR  136  // bf16 probs

__global__ void __launch_bounds__(256)
attention_kernel(const float* __restrict__ Q, const float* __restrict__ Kb,
                 const float* __restrict__ V, float* __restrict__ O) {
  __shared__ __attribute__((aligned(16))) bf16 Qs[QB * QKSTR];
  __shared__ __attribute__((aligned(16))) bf16 Ks[KB * QKSTR];
  __shared__ __attribute__((aligned(16))) bf16 Vs[HD_ * VSTR];
  __shared__ __attribute__((aligned(16))) float Sc[QB * SSTR];
  __shared__ __attribute__((aligned(16))) bf16 Ps[QB * PSTR];
  __shared__ float rm[QB], rl[QB], rscl[QB];
  __shared__ float pmax[2 * QB], psum[2 * QB];

  const int tid = threadIdx.x, lane = tid & 31, wave = tid >> 5;
  const int wm = wave & 3, wn = wave >> 2;
  const int lo = lane & 15, ksel = lane >> 4;
  const int q0 = blockIdx.x * QB, h = blockIdx.y, b = blockIdx.z;
  const size_t baseQ = ((size_t)b * S_ + q0) * D_ + (size_t)h * HD_;
  const float qk_scale = 0.125f;  // 1/sqrt(HD)

  // Q tile (128 x 64 f32 -> bf16)
  for (int idx = tid; idx < QB * HD_ / 4; idx += 256) {
    int r = idx >> 4, c4 = idx & 15;
    const float4 v = *(const float4*)(Q + baseQ + (size_t)r * D_ + c4 * 4);
    v4bf o = {(bf16)v.x, (bf16)v.y, (bf16)v.z, (bf16)v.w};
    *(v4bf*)&Qs[r * QKSTR + c4 * 4] = o;
  }
  for (int r = tid; r < QB; r += 256) { rm[r] = -3.0e38f; rl[r] = 0.0f; }

  v8f acc[2][2];
#pragma unroll
  for (int i = 0; i < 2; ++i)
#pragma unroll
    for (int j = 0; j < 2; ++j) acc[i][j] = zero_v8f();

  for (int kb = 0; kb < S_; kb += KB) {
    __syncthreads();  // previous iteration fully consumed K/V/Sc/Ps
    const size_t baseK = ((size_t)b * S_ + kb) * D_ + (size_t)h * HD_;
    for (int idx = tid; idx < KB * HD_ / 4; idx += 256) {
      int r = idx >> 4, c4 = idx & 15;
      const float4 v = *(const float4*)(Kb + baseK + (size_t)r * D_ + c4 * 4);
      v4bf o = {(bf16)v.x, (bf16)v.y, (bf16)v.z, (bf16)v.w};
      *(v4bf*)&Ks[r * QKSTR + c4 * 4] = o;
      const float4 w = *(const float4*)(V + baseK + (size_t)r * D_ + c4 * 4);
      Vs[(c4 * 4 + 0) * VSTR + r] = (bf16)w.x;   // transposed: Vs[d][key]
      Vs[(c4 * 4 + 1) * VSTR + r] = (bf16)w.y;
      Vs[(c4 * 4 + 2) * VSTR + r] = (bf16)w.z;
      Vs[(c4 * 4 + 3) * VSTR + r] = (bf16)w.w;
    }
    __syncthreads();

    // ---- S = Q @ K^T (each wave: 32q x 64k = 2x4 tiles) ----
    v8f sc[2][4];
#pragma unroll
    for (int i = 0; i < 2; ++i)
#pragma unroll
      for (int j = 0; j < 4; ++j) sc[i][j] = zero_v8f();
#pragma unroll
    for (int kk = 0; kk < HD_; kk += 32) {
      v16bf aq[2], bk[4];
#pragma unroll
      for (int i = 0; i < 2; ++i)
        aq[i] = load_frag(&Qs[(wm * 32 + i * 16) * QKSTR + kk], lo, ksel, QKSTR);
#pragma unroll
      for (int j = 0; j < 4; ++j)
        bk[j] = load_frag(&Ks[(wn * 64 + j * 16) * QKSTR + kk], lo, ksel, QKSTR);
#pragma unroll
      for (int i = 0; i < 2; ++i)
#pragma unroll
        for (int j = 0; j < 4; ++j) sc[i][j] = wmma_bf16(aq[i], bk[j], sc[i][j]);
    }
#pragma unroll
    for (int i = 0; i < 2; ++i)
#pragma unroll
      for (int j = 0; j < 4; ++j)
#pragma unroll
        for (int ii = 0; ii < 8; ++ii)
          Sc[(wm * 32 + i * 16 + ii + 8 * ksel) * SSTR + wn * 64 + j * 16 + lo] =
              sc[i][j][ii] * qk_scale;
    __syncthreads();

    // ---- online softmax (2 threads per row) ----
    {
      const int r = tid >> 1, half = tid & 1;
      const float* srow = &Sc[r * SSTR + half * 64];
      float mx = -3.0e38f;
      for (int c = 0; c < 64; ++c) mx = fmaxf(mx, srow[c]);
      pmax[tid] = mx;
    }
    __syncthreads();
    if ((tid & 1) == 0) {
      const int r = tid >> 1;
      const float mnew = fmaxf(rm[r], fmaxf(pmax[tid], pmax[tid + 1]));
      rscl[r] = __expf(rm[r] - mnew);
      rm[r] = mnew;
    }
    __syncthreads();
    {
      const int r = tid >> 1, half = tid & 1;
      const float m = rm[r];
      float lsum = 0.0f;
      for (int c = 0; c < 64; ++c) {
        const float p = __expf(Sc[r * SSTR + half * 64 + c] - m);
        Ps[r * PSTR + half * 64 + c] = (bf16)p;
        lsum += p;
      }
      psum[tid] = lsum;
    }
    __syncthreads();
    if ((tid & 1) == 0) {
      const int r = tid >> 1;
      rl[r] = rl[r] * rscl[r] + psum[tid] + psum[tid + 1];
    }
    __syncthreads();

    // ---- rescale running O, then O += P @ V (wave: 32q x 32d = 2x2 tiles) ----
#pragma unroll
    for (int i = 0; i < 2; ++i)
#pragma unroll
      for (int ii = 0; ii < 8; ++ii) {
        const float f = rscl[wm * 32 + i * 16 + ii + 8 * ksel];
        acc[i][0][ii] *= f;
        acc[i][1][ii] *= f;
      }
#pragma unroll
    for (int kc = 0; kc < KB; kc += 32) {
      v16bf ap[2], bv[2];
#pragma unroll
      for (int i = 0; i < 2; ++i)
        ap[i] = load_frag(&Ps[(wm * 32 + i * 16) * PSTR + kc], lo, ksel, PSTR);
#pragma unroll
      for (int j = 0; j < 2; ++j)
        bv[j] = load_frag(&Vs[(wn * 32 + j * 16) * VSTR + kc], lo, ksel, VSTR);
#pragma unroll
      for (int i = 0; i < 2; ++i)
#pragma unroll
        for (int j = 0; j < 2; ++j) acc[i][j] = wmma_bf16(ap[i], bv[j], acc[i][j]);
    }
  }

  // ---- normalize & write [T, D] with col = h*64 + d ----
#pragma unroll
  for (int i = 0; i < 2; ++i)
#pragma unroll
    for (int j = 0; j < 2; ++j)
#pragma unroll
      for (int ii = 0; ii < 8; ++ii) {
        const int row = wm * 32 + i * 16 + ii + 8 * ksel;
        O[((size_t)b * S_ + q0 + row) * D_ + (size_t)h * HD_ + wn * 32 + j * 16 + lo] =
            acc[i][j][ii] / rl[row];
      }
}

// ================= residual add + LayerNorm (one block per token) ============
__global__ void __launch_bounds__(256)
add_ln_kernel(const float* __restrict__ a, const float* __restrict__ r,
              const float* __restrict__ g, const float* __restrict__ bta,
              float* __restrict__ out) {
  __shared__ float red[256];
  const int row = blockIdx.x, tid = threadIdx.x;
  const float* pa = a + (size_t)row * D_;
  const float* pr = r + (size_t)row * D_;
  float v[4];
  float sum = 0.0f;
#pragma unroll
  for (int i = 0; i < 4; ++i) {
    const int c = tid + i * 256;
    v[i] = pa[c] + pr[c];
    sum += v[i];
  }
  red[tid] = sum;
  __syncthreads();
  for (int s2 = 128; s2 > 0; s2 >>= 1) {
    if (tid < s2) red[tid] += red[tid + s2];
    __syncthreads();
  }
  const float mu = red[0] / (float)D_;
  __syncthreads();
  float vs = 0.0f;
#pragma unroll
  for (int i = 0; i < 4; ++i) { const float d = v[i] - mu; vs += d * d; }
  red[tid] = vs;
  __syncthreads();
  for (int s2 = 128; s2 > 0; s2 >>= 1) {
    if (tid < s2) red[tid] += red[tid + s2];
    __syncthreads();
  }
  const float inv = rsqrtf(red[0] / (float)D_ + EPS_);
#pragma unroll
  for (int i = 0; i < 4; ++i) {
    const int c = tid + i * 256;
    out[(size_t)row * D_ + c] = (v[i] - mu) * inv * g[c] + bta[c];
  }
}

// ================= gate: softmax over E, top-2, aux-loss stats ===============
__global__ void __launch_bounds__(256)
gate_kernel(const float* __restrict__ xf, const float* __restrict__ gW,
            const float* __restrict__ gb, float* __restrict__ wfull,
            float* __restrict__ stats) {
  __shared__ float red[256];
  __shared__ float logits[E_];
  const int t = blockIdx.x, tid = threadIdx.x;
  float a[E_];
#pragma unroll
  for (int e = 0; e < E_; ++e) a[e] = 0.0f;
  for (int d = tid; d < D_; d += 256) {
    const float xv = xf[(size_t)t * D_ + d];
    const float* wr = gW + (size_t)d * E_;
#pragma unroll
    for (int e = 0; e < E_; ++e) a[e] += xv * wr[e];
  }
  for (int e = 0; e < E_; ++e) {
    red[tid] = a[e];
    __syncthreads();
    for (int s2 = 128; s2 > 0; s2 >>= 1) {
      if (tid < s2) red[tid] += red[tid + s2];
      __syncthreads();
    }
    if (tid == 0) logits[e] = red[0] + gb[e];
    __syncthreads();
  }
  if (tid == 0) {
    float mx = -3.0e38f;
    for (int e = 0; e < E_; ++e) mx = fmaxf(mx, logits[e]);
    float p[E_], se = 0.0f;
    for (int e = 0; e < E_; ++e) { p[e] = __expf(logits[e] - mx); se += p[e]; }
    for (int e = 0; e < E_; ++e) p[e] /= se;
    int i1 = 0;
    for (int e = 1; e < E_; ++e) if (p[e] > p[i1]) i1 = e;
    int i2 = (i1 == 0) ? 1 : 0;
    for (int e = 0; e < E_; ++e) if (e != i1 && p[e] > p[i2]) i2 = e;
    const float s12 = p[i1] + p[i2];
    for (int e = 0; e < E_; ++e) wfull[(size_t)t * E_ + e] = 0.0f;
    wfull[(size_t)t * E_ + i1] = p[i1] / s12;
    wfull[(size_t)t * E_ + i2] = p[i2] / s12;
    atomicAdd(&stats[i1], 1.0f);
    atomicAdd(&stats[i2], 1.0f);
    for (int e = 0; e < E_; ++e) atomicAdd(&stats[E_ + e], p[e]);
  }
}

__global__ void zero_kernel(float* __restrict__ p, size_t n) {
  const size_t i = (size_t)blockIdx.x * 256 + threadIdx.x;
  if (i < n) p[i] = 0.0f;
}

__global__ void aux_kernel(const float* __restrict__ stats, float* __restrict__ out) {
  if (threadIdx.x == 0 && blockIdx.x == 0) {
    float s = 0.0f;
    for (int e = 0; e < E_; ++e)
      s += (stats[e] / (float)(T_ * KTOP_)) * (stats[E_ + e] / (float)T_);
    out[0] = (float)E_ * s;
  }
}

// ================= host-side orchestration ===================================
extern "C" void kernel_launch(void* const* d_in, const int* in_sizes, int n_in,
                              void* d_out, int out_size, void* d_ws, size_t ws_size,
                              hipStream_t stream) {
  (void)in_sizes; (void)n_in; (void)out_size; (void)ws_size;
  const float* x     = (const float*)d_in[0];
  const float* Wq    = (const float*)d_in[1];
  const float* bq    = (const float*)d_in[2];
  const float* Wk    = (const float*)d_in[3];
  const float* bk    = (const float*)d_in[4];
  const float* Wv    = (const float*)d_in[5];
  const float* bv    = (const float*)d_in[6];
  const float* Wo    = (const float*)d_in[7];
  const float* bo    = (const float*)d_in[8];
  const float* g1    = (const float*)d_in[9];
  const float* beta1 = (const float*)d_in[10];
  const float* gateW = (const float*)d_in[11];
  const float* gateb = (const float*)d_in[12];
  const float* eW1   = (const float*)d_in[13];
  const float* eb1   = (const float*)d_in[14];
  const float* eW2   = (const float*)d_in[15];
  const float* eb2   = (const float*)d_in[16];
  const float* g2    = (const float*)d_in[17];
  const float* beta2 = (const float*)d_in[18];
  float* out = (float*)d_out;

  char* ws = (char*)d_ws;
  size_t off = 0;
  auto take = [&](size_t nelem) {
    float* p = (float*)(ws + off);
    off += ((nelem * sizeof(float) + 255) / 256) * 256;
    return p;
  };
  float* q     = take((size_t)T_ * D_);
  float* kbuf  = take((size_t)T_ * D_);
  float* vbuf  = take((size_t)T_ * D_);
  float* attn  = take((size_t)T_ * D_);
  float* proj  = take((size_t)T_ * D_);
  float* x1    = take((size_t)T_ * D_);
  float* hbuf  = take((size_t)T_ * F_);
  float* moe   = take((size_t)T_ * D_);
  float* wfull = take((size_t)T_ * E_);
  float* stats = take(2 * E_);

  const dim3 blk(256);
  const dim3 gproj(D_ / BN, T_ / BM);  // N=1024 -> 8, M=8192 -> 64

  // ---- attention ----
  gemm_kernel<0><<<gproj, blk, 0, stream>>>(x, Wq, bq, q,    T_, D_, D_, nullptr, 0);
  gemm_kernel<0><<<gproj, blk, 0, stream>>>(x, Wk, bk, kbuf, T_, D_, D_, nullptr, 0);
  gemm_kernel<0><<<gproj, blk, 0, stream>>>(x, Wv, bv, vbuf, T_, D_, D_, nullptr, 0);
  const dim3 gattn(S_ / QB, H_, B_);
  attention_kernel<<<gattn, blk, 0, stream>>>(q, kbuf, vbuf, attn);
  gemm_kernel<0><<<gproj, blk, 0, stream>>>(attn, Wo, bo, proj, T_, D_, D_, nullptr, 0);
  add_ln_kernel<<<T_, blk, 0, stream>>>(x, proj, g1, beta1, x1);

  // ---- MoE ----
  const size_t nmoe = (size_t)T_ * D_;
  zero_kernel<<<(unsigned)((nmoe + 255) / 256), blk, 0, stream>>>(moe, nmoe);
  zero_kernel<<<1, blk, 0, stream>>>(stats, 2 * E_);
  gate_kernel<<<T_, blk, 0, stream>>>(x1, gateW, gateb, wfull, stats);

  const dim3 gup(F_ / BN, T_ / BM);    // 32 x 64
  const dim3 gdn(D_ / BN, T_ / BM);    // 8 x 64
  for (int e = 0; e < E_; ++e) {
    // h = gelu(x1 @ eW1[e] + eb1[e]); tiles with all-zero top-2 weight skipped
    gemm_kernel<1><<<gup, blk, 0, stream>>>(x1, eW1 + (size_t)e * D_ * F_,
                                            eb1 + (size_t)e * F_, hbuf,
                                            T_, F_, D_, wfull + e, E_);
    // moe += w[:,e] * (h @ eW2[e] + eb2[e])
    gemm_kernel<2><<<gdn, blk, 0, stream>>>(hbuf, eW2 + (size_t)e * F_ * D_,
                                            eb2 + (size_t)e * D_, moe,
                                            T_, D_, F_, wfull + e, E_);
  }

  // ---- final LN + aux loss ----
  add_ln_kernel<<<T_, blk, 0, stream>>>(x1, moe, g2, beta2, out);
  aux_kernel<<<1, 1, 0, stream>>>(stats, out + (size_t)T_ * D_);
}
